// VectorQuantizer_55748675502202
// MI455X (gfx1250) — matrix operand order repaired and checked by reading, not compiled, not run on hardware
//
#include <hip/hip_runtime.h>
#include <hip/hip_bf16.h>

typedef __attribute__((ext_vector_type(16))) __bf16 v16bf;
typedef __attribute__((ext_vector_type(8)))  float  v8f;

#define NUM_CODES 8192
#define DIM       256
#define BT        32768          // 16 * 2048 rows
#define LDS_STRIDE 272           // 256 + 16 bf16 pad to spread LDS banks

// ---------------------------------------------------------------------------
// Kernel 1: ||e||^2 per codebook row.  One wave32 per code row.
// ---------------------------------------------------------------------------
__global__ __launch_bounds__(256) void vq_norms_kernel(const float* __restrict__ cb,
                                                       float* __restrict__ norms) {
    const int tid  = threadIdx.x;
    const int wave = tid >> 5;
    const int lane = tid & 31;
    const int code = blockIdx.x * 8 + wave;
    const float* row = cb + (size_t)code * DIM + lane * 8;
    float s = 0.f;
#pragma unroll
    for (int i = 0; i < 8; i += 4) {
        float4 v = *(const float4*)(row + i);
        s += v.x * v.x + v.y * v.y + v.z * v.z + v.w * v.w;
    }
#pragma unroll
    for (int off = 16; off >= 1; off >>= 1) s += __shfl_xor(s, off, 32);
    if (lane == 0) norms[code] = s;
}

// ---------------------------------------------------------------------------
// Kernel 2: fused distance GEMM (split-bf16 WMMA) + argmin + gather + loss.
// Block = 256 threads = 8 waves; each wave owns a 16-row A tile (resident in
// VGPRs as bf16 hi/lo); block sweeps all 8192 codes in 16-code N tiles staged
// through LDS.
// ---------------------------------------------------------------------------
__global__ __launch_bounds__(256) void vq_main_kernel(const float* __restrict__ x,
                                                      const float* __restrict__ cb,
                                                      const float* __restrict__ norms,
                                                      float* __restrict__ out_q,
                                                      float* __restrict__ out_idx,
                                                      float* __restrict__ partial) {
    __shared__ __bf16 sh_hi[16 * LDS_STRIDE];
    __shared__ __bf16 sh_lo[16 * LDS_STRIDE];
    __shared__ float  sh_red[8];

    const int tid  = threadIdx.x;
    const int wave = tid >> 5;
    const int lane = tid & 31;
    const int half = lane >> 4;    // lane group g (WMMA operand halves)
    const int col  = lane & 15;    // N column of B/C; also M row of the A load
    const int m0   = blockIdx.x * 128;

    // ---- Load this wave's 16-row A tile, split fp32 -> bf16 hi + lo.
    // 16-bit A 16x32 layout: lane group g holds K = 8g+[0..7] (elems 0..7)
    // and K = 16+8g+[0..7] (elems 8..15) of row M = lane&15.
    v16bf a_hi[8], a_lo[8];
    {
        const float* xrow = x + (size_t)(m0 + wave * 16 + col) * DIM;
#pragma unroll
        for (int j = 0; j < 8; ++j) {
            const int d0 = 32 * j + 8 * half;
            float4 f0 = *(const float4*)(xrow + d0);
            float4 f1 = *(const float4*)(xrow + d0 + 4);
            float4 f2 = *(const float4*)(xrow + d0 + 16);
            float4 f3 = *(const float4*)(xrow + d0 + 20);
            float fv[16] = {f0.x, f0.y, f0.z, f0.w, f1.x, f1.y, f1.z, f1.w,
                            f2.x, f2.y, f2.z, f2.w, f3.x, f3.y, f3.z, f3.w};
#pragma unroll
            for (int e = 0; e < 16; ++e) {
                __bf16 h = (__bf16)fv[e];
                a_hi[j][e] = h;
                a_lo[j][e] = (__bf16)(fv[e] - (float)h);
            }
        }
    }

    float vmin[8];
    int   vidx[8];
#pragma unroll
    for (int v = 0; v < 8; ++v) { vmin[v] = 3.4e38f; vidx[v] = 0; }

    const int sr = tid >> 4;   // staging: code row within tile
    const int sc = tid & 15;   // staging: 16-float chunk within row

    for (int n0 = 0; n0 < NUM_CODES; n0 += 16) {
        // ---- Stage 16-code tile into LDS as bf16 hi/lo (one convert/block).
        __syncthreads();
        {
            const float* crow = cb + (size_t)(n0 + sr) * DIM + sc * 16;
            v16bf hi, lo;
#pragma unroll
            for (int q = 0; q < 4; ++q) {
                float4 f = *(const float4*)(crow + q * 4);
                float fe[4] = {f.x, f.y, f.z, f.w};
#pragma unroll
                for (int e = 0; e < 4; ++e) {
                    __bf16 h = (__bf16)fe[e];
                    hi[q * 4 + e] = h;
                    lo[q * 4 + e] = (__bf16)(fe[e] - (float)h);
                }
            }
            *(v16bf*)(sh_hi + sr * LDS_STRIDE + sc * 16) = hi;
            *(v16bf*)(sh_lo + sr * LDS_STRIDE + sc * 16) = lo;
        }
        __syncthreads();

        const float nrm = norms[n0 + col];

        // ---- 16x16 dot-product tile: 8 K-steps, 3 split-precision WMMAs each.
        // 16-bit B 32x16 layout: lane group g holds K = 8g+[0..7] and K = 16+8g+[0..7] of col N.
        v8f acc = {};
#pragma unroll
        for (int j = 0; j < 8; ++j) {
            const int off = col * LDS_STRIDE + 32 * j + 8 * half;
            typedef __attribute__((ext_vector_type(8))) __bf16 v8bf_t;
            v16bf b_hi = __builtin_shufflevector(*(const v8bf_t*)(sh_hi + off), *(const v8bf_t*)(sh_hi + off + 16), 0, 1, 2, 3, 4, 5, 6, 7, 8, 9, 10, 11, 12, 13, 14, 15);
            v16bf b_lo = __builtin_shufflevector(*(const v8bf_t*)(sh_lo + off), *(const v8bf_t*)(sh_lo + off + 16), 0, 1, 2, 3, 4, 5, 6, 7, 8, 9, 10, 11, 12, 13, 14, 15);
            acc = __builtin_amdgcn_wmma_f32_16x16x32_bf16(false, a_hi[j], false, b_hi,
                                                          (short)0, acc, false, false);
            acc = __builtin_amdgcn_wmma_f32_16x16x32_bf16(false, a_hi[j], false, b_lo,
                                                          (short)0, acc, false, false);
            acc = __builtin_amdgcn_wmma_f32_16x16x32_bf16(false, a_lo[j], false, b_hi,
                                                          (short)0, acc, false, false);
        }

        // ---- dist = ||e||^2 - 2 x.e   (||x||^2 dropped: constant per row).
        // C layout: element v at lane l -> (M = 8*half + v, N = col).
#pragma unroll
        for (int v = 0; v < 8; ++v) {
            float d = nrm - 2.0f * acc[v];
            if (d < vmin[v]) { vmin[v] = d; vidx[v] = n0 + col; }
        }
    }

    // ---- Final cross-lane argmin within each 16-lane half, gather + loss.
    float lsum = 0.f;
#pragma unroll
    for (int v = 0; v < 8; ++v) {
        float m  = vmin[v];
        int   ix = vidx[v];
#pragma unroll
        for (int off = 1; off <= 8; off <<= 1) {
            float om = __shfl_xor(m, off, 32);
            int   oi = __shfl_xor(ix, off, 32);
            if (om < m || (om == m && oi < ix)) { m = om; ix = oi; }
        }
        const int row = m0 + wave * 16 + half * 8 + v;
        if (col == 0) out_idx[row] = (float)ix;

        // 16 lanes of this half cooperatively copy code row `ix` (256 floats)
        // and accumulate (e - x)^2.
        const float* e  = cb + (size_t)ix * DIM + col * 16;
        const float* xi = x + (size_t)row * DIM + col * 16;
        float*       o  = out_q + (size_t)row * DIM + col * 16;
#pragma unroll
        for (int q = 0; q < 4; ++q) {
            float4 ev = *(const float4*)(e + q * 4);
            float4 xv = *(const float4*)(xi + q * 4);
            *(float4*)(o + q * 4) = ev;
            float dx = ev.x - xv.x, dy = ev.y - xv.y;
            float dz = ev.z - xv.z, dw = ev.w - xv.w;
            lsum += dx * dx + dy * dy + dz * dz + dw * dw;
        }
    }

    // ---- Deterministic block loss reduction (fixed tree order, no atomics).
#pragma unroll
    for (int off = 16; off >= 1; off >>= 1) lsum += __shfl_xor(lsum, off, 32);
    if (lane == 0) sh_red[wave] = lsum;
    __syncthreads();
    if (tid == 0) {
        float s = 0.f;
#pragma unroll
        for (int i = 0; i < 8; ++i) s += sh_red[i];
        partial[blockIdx.x] = s;
    }
}

// ---------------------------------------------------------------------------
// Kernel 3: fixed-order final loss reduction.
// loss = q_latent + 0.25 * e_latent = 1.25 * mean((quantized - x)^2)
// ---------------------------------------------------------------------------
__global__ void vq_loss_kernel(const float* __restrict__ partial,
                               float* __restrict__ out_loss) {
    if (threadIdx.x == 0) {
        float s = 0.f;
        for (int i = 0; i < 256; ++i) s += partial[i];
        *out_loss = 1.25f * s / 8388608.0f;
    }
}

// ---------------------------------------------------------------------------
extern "C" void kernel_launch(void* const* d_in, const int* in_sizes, int n_in,
                              void* d_out, int out_size, void* d_ws, size_t ws_size,
                              hipStream_t stream) {
    const float* x  = (const float*)d_in[0];   // inputs  [16,2048,256] f32
    const float* cb = (const float*)d_in[1];   // codebook [8192,256]  f32

    float* out      = (float*)d_out;
    float* out_q    = out;                          // 8388608 floats
    float* out_idx  = out + (size_t)BT * DIM;       // 32768 floats (idx values)
    float* out_loss = out_idx + BT;                 // 1 float

    float* norms   = (float*)d_ws;                  // 8192 floats
    float* partial = norms + NUM_CODES;             // 256 floats

    vq_norms_kernel<<<NUM_CODES / 8, 256, 0, stream>>>(cb, norms);
    vq_main_kernel<<<BT / 128, 256, 0, stream>>>(x, cb, norms, out_q, out_idx, partial);
    vq_loss_kernel<<<1, 64, 0, stream>>>(partial, out_loss);
}
